// SCCNNComplex_58703613001889
// MI455X (gfx1250) — matrix operand order repaired and checked by reading, not compiled, not run on hardware
//
#include <hip/hip_runtime.h>

// ---------------------------------------------------------------------------
// CDNA5 (gfx1250) SCCNN forward — all heavy math via v_wmma_f32_16x16x32_bf16
//  * A staged f32->bf16 through LDS (coalesced b128 loads, packed b128 stores)
//  * B pre-converted once per GEMM to bf16 [n][Kpad]; fragments loaded straight
//    from global as contiguous 16B chunks (L2-resident panel)
//  * split-K with deterministic fixed-order reduction (no float atomics)
// ---------------------------------------------------------------------------

typedef __bf16 v16bf __attribute__((ext_vector_type(16)));
typedef __bf16 v8bf  __attribute__((ext_vector_type(8)));
typedef float  v8f   __attribute__((ext_vector_type(8)));

#define TILE_M 64
#define TILE_K 32
#define LDSK   40   // padded K stride (bf16); 80B rows keep 16B alignment, spread banks

// C[m][n] = sum_t sum_k A_t[m][k] * B_t[k][n]   (N fixed = 128)
// TRANS_A: A_t accessed as A_t[k*lda + m]  (C = A^T @ B)
// Bbf: bf16, layout [tap][n=128][Kpad] (k contiguous, zero-padded to Kpad)
// gridDim.y == 1 : write C directly (+bias). Else: write per-split partials.
template <bool TRANS_A>
__global__ __launch_bounds__(256)
void gemm_n128_bf16(const float* __restrict__ A,
                    const __bf16* __restrict__ Bbf,
                    float* __restrict__ Cpart,
                    float* __restrict__ Cfinal,
                    const float* __restrict__ bias,
                    int M, int K, int lda, int Kpad,
                    int stepsPerTap, int stepsPerSplit, int totalSteps,
                    long aTapStride)
{
    __shared__ __align__(16) __bf16 As[TILE_M * LDSK];   // 5 KB

    const int tid   = threadIdx.x;
    const int lane  = tid & 31;
    const int wave  = tid >> 5;
    const int waveM = wave >> 2;   // 0..1 -> 32-row strip
    const int waveN = wave & 3;    // 0..3 -> 32-col strip
    const int m0    = blockIdx.x * TILE_M;
    const bool fullM = (m0 + TILE_M) <= M;

    const int ml = lane & 15;      // m (A) / n (B) within 16
    const int kh = lane >> 4;      // lane-half -> K interleave group

    v8f acc[2][2];
#pragma unroll
    for (int i = 0; i < 2; ++i)
#pragma unroll
        for (int j = 0; j < 2; ++j)
#pragma unroll
            for (int e = 0; e < 8; ++e) acc[i][j][e] = 0.0f;

    const int sBegin = blockIdx.y * stepsPerSplit;
    int sEnd = sBegin + stepsPerSplit;
    if (sEnd > totalSteps) sEnd = totalSteps;

    int tap = sBegin / stepsPerTap;              // one division, then inc/wrap
    int ks  = sBegin - tap * stepsPerTap;

    for (int st = sBegin; st < sEnd; ++st) {
        const int k0 = ks * TILE_K;
        const float*  At = A   + (long)tap * aTapStride;
        const __bf16* Bt = Bbf + (long)tap * 128 * Kpad;
        const bool fullK = (k0 + TILE_K) <= K;

        // ---- stage A tile (TILE_M x TILE_K) f32 -> bf16 into LDS ----
        if (!TRANS_A) {
            if (fullM && fullK) {
                // thread: one row-segment of 8 floats = 2x b128 loads, 1x b128 store
                const int r = tid >> 2, cs = (tid & 3) * 8;
                const float4* p = (const float4*)(At + (long)(m0 + r) * lda + (k0 + cs));
                float4 u = p[0], v = p[1];
                v8bf w;
                w[0] = (__bf16)u.x; w[1] = (__bf16)u.y; w[2] = (__bf16)u.z; w[3] = (__bf16)u.w;
                w[4] = (__bf16)v.x; w[5] = (__bf16)v.y; w[6] = (__bf16)v.z; w[7] = (__bf16)v.w;
                *(v8bf*)&As[r * LDSK + cs] = w;
            } else {
#pragma unroll
                for (int i = 0; i < 8; ++i) {
                    int idx = tid + i * 256;
                    int r = idx >> 5, c = idx & 31;
                    int gm = m0 + r, gk = k0 + c;
                    float v = (gm < M && gk < K) ? At[(long)gm * lda + gk] : 0.0f;
                    As[r * LDSK + c] = (__bf16)v;
                }
            }
        } else {
            if (fullM && fullK) {
                // thread: 8 consecutive m at one k (coalesced along m)
                const int kk = tid >> 3, ms = (tid & 7) * 8;
                const float4* p = (const float4*)(At + (long)(k0 + kk) * lda + (m0 + ms));
                float4 u = p[0], v = p[1];
                As[(ms + 0) * LDSK + kk] = (__bf16)u.x;
                As[(ms + 1) * LDSK + kk] = (__bf16)u.y;
                As[(ms + 2) * LDSK + kk] = (__bf16)u.z;
                As[(ms + 3) * LDSK + kk] = (__bf16)u.w;
                As[(ms + 4) * LDSK + kk] = (__bf16)v.x;
                As[(ms + 5) * LDSK + kk] = (__bf16)v.y;
                As[(ms + 6) * LDSK + kk] = (__bf16)v.z;
                As[(ms + 7) * LDSK + kk] = (__bf16)v.w;
            } else {
#pragma unroll
                for (int i = 0; i < 8; ++i) {
                    int idx = tid + i * 256;
                    int c = idx >> 6, r = idx & 63;
                    int gm = m0 + r, gk = k0 + c;
                    float v = (gm < M && gk < K) ? At[(long)gk * lda + gm] : 0.0f;
                    As[r * LDSK + c] = (__bf16)v;
                }
            }
        }
        __syncthreads();

        // ---- fragments ----
        v16bf afrag[2], bfrag[2];
#pragma unroll
        for (int i = 0; i < 2; ++i) {
            // A 16x32 bf16: lane half h -> K = {h*8..h*8+7, 16+h*8..16+h*8+7}
            int row = waveM * 32 + i * 16 + ml;
            v8bf lo = *(const v8bf*)&As[row * LDSK + kh * 8];
            v8bf hi = *(const v8bf*)&As[row * LDSK + 16 + kh * 8];
#pragma unroll
            for (int e = 0; e < 8; ++e) { afrag[i][e] = lo[e]; afrag[i][e + 8] = hi[e]; }
        }
#pragma unroll
        for (int j = 0; j < 2; ++j) {
            // B 32x16 bf16: lane half h -> K = {h*8..h*8+7, 16+h*8..16+h*8+7}
            const __bf16* bp = Bt + (long)(waveN * 32 + j * 16 + ml) * Kpad + k0 + kh * 8;
            v8bf lo = *(const v8bf*)bp;
            v8bf hi = *(const v8bf*)(bp + 16);
#pragma unroll
            for (int e = 0; e < 8; ++e) { bfrag[j][e] = lo[e]; bfrag[j][e + 8] = hi[e]; }
        }
#pragma unroll
        for (int i = 0; i < 2; ++i)
#pragma unroll
            for (int j = 0; j < 2; ++j)
                acc[i][j] = __builtin_amdgcn_wmma_f32_16x16x32_bf16(
                    false, afrag[i], false, bfrag[j], (short)0, acc[i][j], false, false);
        __syncthreads();

        if (++ks == stepsPerTap) { ks = 0; ++tap; }
    }

    // ---- store ----
    if (gridDim.y == 1) {
#pragma unroll
        for (int i = 0; i < 2; ++i)
#pragma unroll
            for (int j = 0; j < 2; ++j) {
                int gn = waveN * 32 + j * 16 + ml;
                float b = bias ? bias[gn] : 0.0f;
#pragma unroll
                for (int r = 0; r < 8; ++r) {
                    int gm = m0 + waveM * 32 + i * 16 + kh * 8 + r;
                    if (gm < M) Cfinal[(long)gm * 128 + gn] = acc[i][j][r] + b;
                }
            }
    } else {
        float* Cp = Cpart + (long)blockIdx.y * (long)M * 128;
#pragma unroll
        for (int i = 0; i < 2; ++i)
#pragma unroll
            for (int j = 0; j < 2; ++j) {
                int gn = waveN * 32 + j * 16 + ml;
#pragma unroll
                for (int r = 0; r < 8; ++r) {
                    int gm = m0 + waveM * 32 + i * 16 + kh * 8 + r;
                    if (gm < M) Cp[(long)gm * 128 + gn] = acc[i][j][r];
                }
            }
    }
}

// B (f32, [tap][K][128]) -> bf16 [tap][n=128][Kpad], zero-padded tail.
__global__ void convert_b_kernel(const float* __restrict__ B, __bf16* __restrict__ Bbf,
                                 int K, int Kpad, long bTapStride, long total)
{
    long idx = (long)blockIdx.x * 256 + threadIdx.x;
    if (idx >= total) return;
    int  n = (int)(idx & 127);          // lanes contiguous along n -> coalesced reads
    long r = idx >> 7;                  // [tap][k]
    int  k = (int)(r % Kpad);
    int  t = (int)(r / Kpad);
    float v = (k < K) ? B[(long)t * bTapStride + (long)k * 128 + n] : 0.0f;
    Bbf[((long)t * 128 + n) * (long)Kpad + k] = (__bf16)v;
}

// Sum split-K partials in fixed order (+ optional bias). float4 vectorized.
__global__ void reduce_bias_kernel(const float* __restrict__ part,
                                   float* __restrict__ C,
                                   long MN4, int S, const float* __restrict__ bias)
{
    long i4 = (long)blockIdx.x * 256 + threadIdx.x;
    if (i4 >= MN4) return;
    const float4* p = (const float4*)part;
    float4 s = p[i4];
    for (int i = 1; i < S; ++i) {
        float4 q = p[(long)i * MN4 + i4];
        s.x += q.x; s.y += q.y; s.z += q.z; s.w += q.w;
    }
    if (bias) {
        float4 b = ((const float4*)bias)[i4 & 31];   // (i4*4)%128 / 4
        s.x += b.x; s.y += b.y; s.z += b.z; s.w += b.w;
    }
    ((float4*)C)[i4] = s;
}

// dst[i][o] = src[o][i]   (128x128): torch Linear weight [out,in] -> [in,out]
__global__ void transpose128_kernel(const float* __restrict__ src, float* __restrict__ dst)
{
    int idx = blockIdx.x * 256 + threadIdx.x;   // 16384
    int o = idx >> 7, i = idx & 127;
    dst[i * 128 + o] = src[o * 128 + i];
}

// src [in=128][out=128][T] -> dst [T][in][out]
__global__ void repack_w_kernel(const float* __restrict__ src, float* __restrict__ dst, int T)
{
    int idx = blockIdx.x * 256 + threadIdx.x;
    if (idx >= 16384 * T) return;
    int t   = idx / 16384;
    int rem = idx - t * 16384;
    int i = rem >> 7, o = rem & 127;
    dst[idx] = src[(i * 128 + o) * T + t];
}

// logits = h0 @ out_w^T + out_b ; sigmoid.  out_w: [2][128]
__global__ void logits_sigmoid_kernel(const float* __restrict__ h0,
                                      const float* __restrict__ w,
                                      const float* __restrict__ b,
                                      float* __restrict__ out, int N)
{
    int n = blockIdx.x * 256 + threadIdx.x;
    if (n >= N) return;
    float a0 = 0.0f, a1 = 0.0f;
    for (int i = 0; i < 128; ++i) {
        float x = h0[(long)n * 128 + i];
        a0 += x * w[i];
        a1 += x * w[128 + i];
    }
    a0 += b[0]; a1 += b[1];
    out[2 * n]     = 1.0f / (1.0f + expf(-a0));
    out[2 * n + 1] = 1.0f / (1.0f + expf(-a1));
}

// ---------------------------------------------------------------------------

static inline void run_gemm(bool transA, const float* A, const float* Bsrc, float* C,
                            int M, int K, int lda, int taps, long aTap, long bTap,
                            const float* bias, float* part, __bf16* bbuf,
                            hipStream_t stream)
{
    const int Kpad = ((K + TILE_K - 1) / TILE_K) * TILE_K;

    long btot = (long)taps * 128 * Kpad;
    convert_b_kernel<<<(btot + 255) / 256, 256, 0, stream>>>(Bsrc, bbuf, K, Kpad, bTap, btot);

    int gridM = (M + TILE_M - 1) / TILE_M;
    int spt   = Kpad / TILE_K;
    int total = taps * spt;
    int S = 768 / gridM;
    if (S < 1) S = 1;
    if (S > 8) S = 8;
    if (S > total) S = total;
    int sps = (total + S - 1) / S;
    S = (total + sps - 1) / sps;   // drop empty splits

    dim3 grid(gridM, S);
    if (transA)
        gemm_n128_bf16<true><<<grid, 256, 0, stream>>>(A, bbuf, part, C, bias,
                                                       M, K, lda, Kpad, spt, sps, total, aTap);
    else
        gemm_n128_bf16<false><<<grid, 256, 0, stream>>>(A, bbuf, part, C, bias,
                                                        M, K, lda, Kpad, spt, sps, total, aTap);
    if (S > 1) {
        long MN4 = (long)M * 128 / 4;
        reduce_bias_kernel<<<(MN4 + 255) / 256, 256, 0, stream>>>(part, C, MN4, S, bias);
    }
}

extern "C" void kernel_launch(void* const* d_in, const int* in_sizes, int n_in,
                              void* d_out, int out_size, void* d_ws, size_t ws_size,
                              hipStream_t stream)
{
    (void)in_sizes; (void)n_in; (void)out_size; (void)ws_size;

    const int N0 = 2000, N1 = 6000, N2 = 4000, C = 128;

    const float* x0    = (const float*)d_in[0];
    const float* x1    = (const float*)d_in[1];
    const float* x2    = (const float*)d_in[2];
    const float* L0    = (const float*)d_in[3];
    const float* L1d   = (const float*)d_in[4];
    const float* L1u   = (const float*)d_in[5];
    const float* L2d   = (const float*)d_in[6];
    const float* L2u   = (const float*)d_in[7];
    const float* B1    = (const float*)d_in[8];   // [N0,N1]
    const float* B2    = (const float*)d_in[9];   // [N1,N2]
    const float* in_w0 = (const float*)d_in[10];
    const float* in_b0 = (const float*)d_in[11];
    const float* in_w1 = (const float*)d_in[12];
    const float* in_b1 = (const float*)d_in[13];
    const float* in_w2 = (const float*)d_in[14];
    const float* in_b2 = (const float*)d_in[15];
    const float* lw[2][3] = {
        { (const float*)d_in[16], (const float*)d_in[17], (const float*)d_in[18] },
        { (const float*)d_in[19], (const float*)d_in[20], (const float*)d_in[21] } };
    const float* out_w = (const float*)d_in[22];
    const float* out_b = (const float*)d_in[23];
    float* out = (float*)d_out;

    // ---- workspace carve (float units) ----
    float* ws = (float*)d_ws;
    size_t off = 0;
    auto carve = [&](size_t n) { float* p = ws + off; off += n; return p; };
    const size_t S0 = (size_t)N0 * C, S1 = (size_t)N1 * C, S2 = (size_t)N2 * C;

    float* h0  = carve(S0);
    float* h1  = carve(S1);
    float* h2  = carve(S2);
    float* Xn  = carve(6 * S0);    // node taps  [tap][N0][C]
    float* Xe  = carve(11 * S1);   // edge taps  [tap][N1][C]
    float* Xf  = carve(8 * S2);    // face taps  [tap][N2][C]
    float* wt0 = carve(16384);
    float* wt1 = carve(16384);
    float* wt2 = carve(16384);
    float* Wr0 = carve(6 * 16384);
    float* Wr1 = carve(11 * 16384);
    float* Wr2 = carve(8 * 16384);
    float* part = carve((size_t)8 * N1 * C);            // split-K partials
    __bf16* bbuf = (__bf16*)carve(385024);              // bf16 B panel, max 128*6016

    // ---- input linears: h = x @ W^T + b ----
    transpose128_kernel<<<64, 256, 0, stream>>>(in_w0, wt0);
    transpose128_kernel<<<64, 256, 0, stream>>>(in_w1, wt1);
    transpose128_kernel<<<64, 256, 0, stream>>>(in_w2, wt2);
    run_gemm(false, x0, wt0, h0, N0, C, C, 1, 0, 0, in_b0, part, bbuf, stream);
    run_gemm(false, x1, wt1, h1, N1, C, C, 1, 0, 0, in_b1, part, bbuf, stream);
    run_gemm(false, x2, wt2, h2, N2, C, C, 1, 0, 0, in_b2, part, bbuf, stream);

    for (int layer = 0; layer < 2; ++layer) {
        repack_w_kernel<<<(6 * 16384 + 255) / 256, 256, 0, stream>>>(lw[layer][0], Wr0, 6);
        repack_w_kernel<<<(11 * 16384 + 255) / 256, 256, 0, stream>>>(lw[layer][1], Wr1, 11);
        repack_w_kernel<<<(8 * 16384 + 255) / 256, 256, 0, stream>>>(lw[layer][2], Wr2, 8);

        // tap 0 = identity
        hipMemcpyAsync(Xn, h0, S0 * sizeof(float), hipMemcpyDeviceToDevice, stream);
        hipMemcpyAsync(Xe, h1, S1 * sizeof(float), hipMemcpyDeviceToDevice, stream);
        hipMemcpyAsync(Xf, h2, S2 * sizeof(float), hipMemcpyDeviceToDevice, stream);

        // ---- node taps: [h0, L0 h0, L0^2 h0, B1 h1, L0 B1 h1, L0^2 B1 h1] ----
        run_gemm(false, L0, Xn,          Xn + 1 * S0, N0, N0, N0, 1, 0, 0, nullptr, part, bbuf, stream);
        run_gemm(false, L0, Xn + 1 * S0, Xn + 2 * S0, N0, N0, N0, 1, 0, 0, nullptr, part, bbuf, stream);
        run_gemm(false, B1, h1,          Xn + 3 * S0, N0, N1, N1, 1, 0, 0, nullptr, part, bbuf, stream);
        run_gemm(false, L0, Xn + 3 * S0, Xn + 4 * S0, N0, N0, N0, 1, 0, 0, nullptr, part, bbuf, stream);
        run_gemm(false, L0, Xn + 4 * S0, Xn + 5 * S0, N0, N0, N0, 1, 0, 0, nullptr, part, bbuf, stream);

        // ---- edge taps ----
        run_gemm(false, L1d, Xe,          Xe + 1 * S1, N1, N1, N1, 1, 0, 0, nullptr, part, bbuf, stream);
        run_gemm(false, L1d, Xe + 1 * S1, Xe + 2 * S1, N1, N1, N1, 1, 0, 0, nullptr, part, bbuf, stream);
        run_gemm(false, L1u, Xe,          Xe + 3 * S1, N1, N1, N1, 1, 0, 0, nullptr, part, bbuf, stream);
        run_gemm(false, L1u, Xe + 3 * S1, Xe + 4 * S1, N1, N1, N1, 1, 0, 0, nullptr, part, bbuf, stream);
        run_gemm(true,  B1,  h0,          Xe + 5 * S1, N1, N0, N1, 1, 0, 0, nullptr, part, bbuf, stream);
        run_gemm(false, L1d, Xe + 5 * S1, Xe + 6 * S1, N1, N1, N1, 1, 0, 0, nullptr, part, bbuf, stream);
        run_gemm(false, L1d, Xe + 6 * S1, Xe + 7 * S1, N1, N1, N1, 1, 0, 0, nullptr, part, bbuf, stream);
        run_gemm(false, B2,  h2,          Xe + 8 * S1, N1, N2, N2, 1, 0, 0, nullptr, part, bbuf, stream);
        run_gemm(false, L1u, Xe + 8 * S1, Xe + 9 * S1, N1, N1, N1, 1, 0, 0, nullptr, part, bbuf, stream);
        run_gemm(false, L1u, Xe + 9 * S1, Xe + 10 * S1, N1, N1, N1, 1, 0, 0, nullptr, part, bbuf, stream);

        // ---- face taps ----
        run_gemm(false, L2d, Xf,          Xf + 1 * S2, N2, N2, N2, 1, 0, 0, nullptr, part, bbuf, stream);
        run_gemm(false, L2d, Xf + 1 * S2, Xf + 2 * S2, N2, N2, N2, 1, 0, 0, nullptr, part, bbuf, stream);
        run_gemm(false, L2u, Xf,          Xf + 3 * S2, N2, N2, N2, 1, 0, 0, nullptr, part, bbuf, stream);
        run_gemm(false, L2u, Xf + 3 * S2, Xf + 4 * S2, N2, N2, N2, 1, 0, 0, nullptr, part, bbuf, stream);
        run_gemm(true,  B2,  h1,          Xf + 5 * S2, N2, N1, N2, 1, 0, 0, nullptr, part, bbuf, stream);
        run_gemm(false, L2d, Xf + 5 * S2, Xf + 6 * S2, N2, N2, N2, 1, 0, 0, nullptr, part, bbuf, stream);
        run_gemm(false, L2d, Xf + 6 * S2, Xf + 7 * S2, N2, N2, N2, 1, 0, 0, nullptr, part, bbuf, stream);

        // ---- tap contractions (einsum "nik,iok->no") -> overwrite h ----
        run_gemm(false, Xn, Wr0, h0, N0, C, C, 6,  (long)S0, 16384, nullptr, part, bbuf, stream);
        run_gemm(false, Xe, Wr1, h1, N1, C, C, 11, (long)S1, 16384, nullptr, part, bbuf, stream);
        run_gemm(false, Xf, Wr2, h2, N2, C, C, 8,  (long)S2, 16384, nullptr, part, bbuf, stream);
    }

    // ---- output head: sigmoid(h0 @ out_w^T + out_b) ----
    logits_sigmoid_kernel<<<(N0 + 255) / 256, 256, 0, stream>>>(h0, out_w, out_b, out, N0);
}
